// MultiHeadSelfAttention_4647154614687
// MI455X (gfx1250) — matrix operand order repaired and checked by reading, not compiled, not run on hardware
//
#include <hip/hip_runtime.h>
#include <hip/hip_bf16.h>

// ---------------------------------------------------------------------------
// Windowed MHSA for MI455X (gfx1250): bf16 WMMA everywhere, flash-attention
// core, wave32 / 8-wave blocks. GEMMs: 32x64 tile per wave (8 WMMA / k-step).
// Attention: 16 query rows per wave, 64-key super-tiles, online softmax.
// ---------------------------------------------------------------------------

typedef __attribute__((ext_vector_type(16))) __bf16 v16bf;
typedef __attribute__((ext_vector_type(8)))  float  v8f;

#define C_DIM   256
#define H_NUM   8
#define S_NUM   512
#define B_NUM   64
#define D_HEAD  32
#define NTOK    (B_NUM * S_NUM)          /* 32768 */
#define QSCALE  0.35355339059327373f     /* 8^-0.5 */

union AFrag { v16bf v; uint4 u[2]; };
union Acc   { v8f   v; float f[8]; };

#define WMMA_BF16(A, B, C) \
    __builtin_amdgcn_wmma_f32_16x16x32_bf16(false, (A), false, (B), (short)0, (C), false, false)

// A-fragment (16x32 bf16, MxK): lane&15 -> M, e -> K = e + (e>=8?8:0) + (lane>=16?8:0)
__device__ inline v16bf load_a16(const __bf16* base, int row0, int k0, int ld) {
    int lane = threadIdx.x & 31;
    int hi   = (lane >> 4) << 3;
    const __bf16* p = base + (size_t)(row0 + (lane & 15)) * ld + k0 + hi;
    AFrag a;
    a.u[0] = *(const uint4*)p;          // e0..7  -> K = hi..hi+7
    a.u[1] = *(const uint4*)(p + 16);   // e8..15 -> K = 16+hi..16+hi+7
    return a.v;
}

// B-fragment (32x16 bf16, KxN): lane&15 -> N, e -> K = e + (e>=8?8:0) + (lane>=16?8:0) (row-major [K x N] source)
__device__ inline v16bf load_b16(const __bf16* base, int k0, int n0, int ld) {
    int lane = threadIdx.x & 31;
    int hi   = (lane >> 4) << 3;
    const __bf16* p = base + (size_t)(k0 + hi) * ld + n0 + (lane & 15);
    AFrag b;
    for (int e = 0; e < 16; ++e)
        b.v[e] = p[(size_t)(e + (e >= 8 ? 8 : 0)) * ld];
    return b.v;
}

// rel_table shape (2T-1=63, L=16, L=16, H=8); bias(i,j,h) = rel[ti-tj+31][li][lj][h]
__device__ inline float rel_bias(const float* rel, int i, int j, int h) {
    int ti = i >> 4, li = i & 15, tj = j >> 4, lj = j & 15;
    return rel[(((size_t)(ti - tj + 31) * 16 + li) * 16 + lj) * 8 + h];
}

// ---------------------------------------------------------------- converters
__global__ void k_cvt_x(const float* __restrict__ x, __bf16* __restrict__ xb, int n) {
    int i = blockIdx.x * blockDim.x + threadIdx.x;
    if (i < n) xb[i] = (__bf16)x[i];
}

// wqkv[k][n] = qkv_w[n][k] (bf16, [256 x 768]); wproj[k][n] = proj_w[n][k] ([256 x 256])
__global__ void k_cvt_w(const float* __restrict__ qkv_w, const float* __restrict__ proj_w,
                        __bf16* __restrict__ wqkv, __bf16* __restrict__ wproj) {
    int i = blockIdx.x * blockDim.x + threadIdx.x;
    if (i < 256 * 768) {
        int k = i / 768, n = i % 768;
        wqkv[i] = (__bf16)qkv_w[n * 256 + k];
    } else {
        int j = i - 256 * 768;
        if (j < 256 * 256) {
            int k = j / 256, n = j % 256;
            wproj[j] = (__bf16)proj_w[n * 256 + k];
        }
    }
}

// ------------------------------------------------- stage 1: QKV GEMM + scatter
// Y[32768 x 768] = xb @ wqkv + bias; 32x64 tile per wave; scatter epilogue to
// q [bh,S,d] (pre-scaled), kT [bh,d,S], v [bh,S,d] in bf16.
__global__ __launch_bounds__(256) void k_qkv(const __bf16* __restrict__ xb,
                                             const __bf16* __restrict__ wqkv,
                                             const float*  __restrict__ qkv_b,
                                             __bf16* __restrict__ qbuf,
                                             __bf16* __restrict__ kT,
                                             __bf16* __restrict__ vbuf) {
    int wave = threadIdx.x >> 5, lane = threadIdx.x & 31;
    int tile = blockIdx.x * 8 + wave;              // 1024 x 12 = 12288 tiles
    int tm = tile / 12, tn = tile % 12;
    int row0 = tm << 5, n0 = tn << 6;

    Acc acc[2][4];
    for (int mi = 0; mi < 2; ++mi)
        for (int ni = 0; ni < 4; ++ni)
            for (int r = 0; r < 8; ++r) acc[mi][ni].f[r] = 0.f;

    for (int kk = 0; kk < 8; ++kk) {
        v16bf a0 = load_a16(xb, row0,      kk * 32, C_DIM);
        v16bf a1 = load_a16(xb, row0 + 16, kk * 32, C_DIM);
        if (kk < 7) {   // gfx1250 global_prefetch_b8 for next k-step A rows
            __builtin_prefetch(xb + (size_t)(row0 + (lane & 15)) * C_DIM + kk * 32 + 32, 0, 3);
            __builtin_prefetch(xb + (size_t)(row0 + 16 + (lane & 15)) * C_DIM + kk * 32 + 32, 0, 3);
        }
        for (int ni = 0; ni < 4; ++ni) {
            v16bf b = load_b16(wqkv, kk * 32, n0 + ni * 16, 3 * C_DIM);
            acc[0][ni].v = WMMA_BF16(a0, b, acc[0][ni].v);
            acc[1][ni].v = WMMA_BF16(a1, b, acc[1][ni].v);
        }
    }

    int hi8 = (lane >> 4) << 3;
    int jl = lane & 15;
    for (int ni = 0; ni < 4; ++ni) {
        int o = n0 + ni * 16 + jl;
        int which = o >> 8;             // 0=q 1=k 2=v (uniform per wave per ni)
        int rem = o & 255;
        int h = rem >> 5, dd = rem & 31;
        float bias = qkv_b[o];
        for (int mi = 0; mi < 2; ++mi) {
            for (int r = 0; r < 8; ++r) {
                int i  = row0 + mi * 16 + r + hi8;   // token id
                int b_ = i >> 9, s = i & 511;
                int bh = b_ * H_NUM + h;
                float val = acc[mi][ni].f[r] + bias;
                if (which == 0)      qbuf[((size_t)bh * S_NUM + s) * D_HEAD + dd] = (__bf16)(val * QSCALE);
                else if (which == 1) kT[((size_t)bh * D_HEAD + dd) * S_NUM + s]   = (__bf16)val;
                else                 vbuf[((size_t)bh * S_NUM + s) * D_HEAD + dd] = (__bf16)val;
            }
        }
    }
}

// ------------------------------------- stage 2: flash attention per (b,h) tile
__global__ __launch_bounds__(256) void k_attn(const __bf16* __restrict__ qbuf,
                                              const __bf16* __restrict__ kT,
                                              const __bf16* __restrict__ vbuf,
                                              const float*  __restrict__ rel,
                                              __bf16* __restrict__ ostage) {
    __shared__ __align__(16) __bf16 plds[8][16][64];   // 16 KB: per-wave P bounce

    int wave = threadIdx.x >> 5, lane = threadIdx.x & 31;
    int bh = blockIdx.x >> 2;            // 512 (b,h) pairs
    int chunk = blockIdx.x & 3;          // 4 chunks of 128 query rows
    int b_ = bh >> 3, h = bh & 7;
    int row0 = chunk * 128 + wave * 16;
    int hi8 = (lane >> 4) << 3;
    int jl  = lane & 15;

    const __bf16* qb = qbuf + (size_t)bh * S_NUM * D_HEAD;
    const __bf16* kb = kT   + (size_t)bh * D_HEAD * S_NUM;
    const __bf16* vb = vbuf + (size_t)bh * S_NUM * D_HEAD;

    v16bf qf = load_a16(qb, row0, 0, D_HEAD);   // Q fragment, constant over key loop

    Acc a0, a1;
    float m[8], l[8];
    for (int r = 0; r < 8; ++r) { a0.f[r] = 0.f; a1.f[r] = 0.f; m[r] = -3.0e38f; l[r] = 0.f; }

    for (int jt = 0; jt < 8; ++jt) {
        int col0 = jt * 64;

        // scores for 64 keys: Q (16x32, K=d) x K^T (32x16) x4
        Acc s[4];
        for (int c = 0; c < 4; ++c) {
            v16bf kf = load_b16(kb, 0, col0 + c * 16, S_NUM);
            for (int r = 0; r < 8; ++r) s[c].f[r] = 0.f;
            s[c].v = WMMA_BF16(qf, kf, s[c].v);
        }

        // relative position bias (rel table is L2-resident: 516 KB)
        for (int r = 0; r < 8; ++r) {
            int i = row0 + r + hi8;
            for (int c = 0; c < 4; ++c)
                s[c].f[r] += rel_bias(rel, i, col0 + c * 16 + jl, h);
        }

        // online softmax over 64 columns; rows live in 16-lane halves
        for (int r = 0; r < 8; ++r) {
            float mt = fmaxf(fmaxf(s[0].f[r], s[1].f[r]), fmaxf(s[2].f[r], s[3].f[r]));
            for (int off = 8; off; off >>= 1) mt = fmaxf(mt, __shfl_xor(mt, off, 32));
            float mnew = fmaxf(m[r], mt);
            float sc = __expf(m[r] - mnew);
            float rs = 0.f;
            int lr = r + hi8;
            for (int c = 0; c < 4; ++c) {
                float e = __expf(s[c].f[r] - mnew);
                rs += e;
                plds[wave][lr][c * 16 + jl] = (__bf16)e;   // stash P as we go
            }
            for (int off = 8; off; off >>= 1) rs += __shfl_xor(rs, off, 32);
            l[r] = l[r] * sc + rs;
            a0.f[r] *= sc; a1.f[r] *= sc;
            m[r] = mnew;
        }

        // O += P (16x64) x V (64x32), two K=32 steps; P refetched in A-layout
        asm volatile("s_wait_dscnt 0" ::: "memory");
        for (int kc = 0; kc < 2; ++kc) {
            AFrag pa;
            const __bf16* lp = &plds[wave][jl][kc * 32 + hi8];
            pa.u[0] = *(const uint4*)lp;
            pa.u[1] = *(const uint4*)(lp + 16);
            v16bf vb0 = load_b16(vb, col0 + kc * 32, 0,  D_HEAD);
            v16bf vb1 = load_b16(vb, col0 + kc * 32, 16, D_HEAD);
            a0.v = WMMA_BF16(pa.v, vb0, a0.v);
            a1.v = WMMA_BF16(pa.v, vb1, a1.v);
        }
    }

    // normalize and stage as [B, S, C] bf16 for the projection GEMM
    for (int r = 0; r < 8; ++r) {
        int i = row0 + r + hi8;
        float inv = 1.0f / l[r];
        size_t base = ((size_t)b_ * S_NUM + i) * C_DIM + h * D_HEAD;
        ostage[base + jl]      = (__bf16)(a0.f[r] * inv);
        ostage[base + 16 + jl] = (__bf16)(a1.f[r] * inv);
    }
}

// ---------------------------------------------------- stage 3: output proj
// out[32768 x 256] = ostage @ wproj + bias; 32x64 tile per wave, f32 output.
__global__ __launch_bounds__(256) void k_proj(const __bf16* __restrict__ ostage,
                                              const __bf16* __restrict__ wproj,
                                              const float*  __restrict__ proj_b,
                                              float* __restrict__ out) {
    int wave = threadIdx.x >> 5, lane = threadIdx.x & 31;
    int tile = blockIdx.x * 8 + wave;          // 1024 x 4 = 4096 tiles
    int tm = tile >> 2, tn = tile & 3;
    int row0 = tm << 5, n0 = tn << 6;

    Acc acc[2][4];
    for (int mi = 0; mi < 2; ++mi)
        for (int ni = 0; ni < 4; ++ni)
            for (int r = 0; r < 8; ++r) acc[mi][ni].f[r] = 0.f;

    for (int kk = 0; kk < 8; ++kk) {
        v16bf a0 = load_a16(ostage, row0,      kk * 32, C_DIM);
        v16bf a1 = load_a16(ostage, row0 + 16, kk * 32, C_DIM);
        if (kk < 7) {
            __builtin_prefetch(ostage + (size_t)(row0 + (lane & 15)) * C_DIM + kk * 32 + 32, 0, 3);
            __builtin_prefetch(ostage + (size_t)(row0 + 16 + (lane & 15)) * C_DIM + kk * 32 + 32, 0, 3);
        }
        for (int ni = 0; ni < 4; ++ni) {
            v16bf b = load_b16(wproj, kk * 32, n0 + ni * 16, C_DIM);
            acc[0][ni].v = WMMA_BF16(a0, b, acc[0][ni].v);
            acc[1][ni].v = WMMA_BF16(a1, b, acc[1][ni].v);
        }
    }

    int hi8 = (lane >> 4) << 3;
    int jl = lane & 15;
    for (int ni = 0; ni < 4; ++ni) {
        int o = n0 + ni * 16 + jl;
        float bias = proj_b[o];
        for (int mi = 0; mi < 2; ++mi)
            for (int r = 0; r < 8; ++r) {
                int i = row0 + mi * 16 + r + hi8;
                out[(size_t)i * C_DIM + o] = acc[mi][ni].f[r] + bias;
            }
    }
}

// ---------------------------------------------------------------------------
extern "C" void kernel_launch(void* const* d_in, const int* in_sizes, int n_in,
                              void* d_out, int out_size, void* d_ws, size_t ws_size,
                              hipStream_t stream) {
    const float* x      = (const float*)d_in[0];
    const float* qkv_w  = (const float*)d_in[1];
    const float* qkv_b  = (const float*)d_in[2];
    const float* proj_w = (const float*)d_in[3];
    const float* proj_b = (const float*)d_in[4];
    const float* rel    = (const float*)d_in[5];
    float* out = (float*)d_out;

    char* ws = (char*)d_ws;
    size_t off = 0;
    __bf16* xb    = (__bf16*)(ws + off); off += (size_t)NTOK * C_DIM * 2;            // 16 MB
    __bf16* wqkv  = (__bf16*)(ws + off); off += (size_t)C_DIM * 3 * C_DIM * 2;       // 384 KB
    __bf16* wproj = (__bf16*)(ws + off); off += (size_t)C_DIM * C_DIM * 2;           // 128 KB
    __bf16* qbuf  = (__bf16*)(ws + off); off += (size_t)B_NUM * H_NUM * S_NUM * D_HEAD * 2;
    __bf16* kT    = (__bf16*)(ws + off); off += (size_t)B_NUM * H_NUM * S_NUM * D_HEAD * 2;
    __bf16* vbuf  = (__bf16*)(ws + off); off += (size_t)B_NUM * H_NUM * S_NUM * D_HEAD * 2;
    __bf16* ost   = (__bf16*)(ws + off); off += (size_t)NTOK * C_DIM * 2;
    (void)ws_size; (void)in_sizes; (void)n_in; (void)out_size;

    int nX = NTOK * C_DIM;                                   // 8388608
    k_cvt_x<<<(nX + 255) / 256, 256, 0, stream>>>(x, xb, nX);
    k_cvt_w<<<(256 * 768 + 256 * 256 + 255) / 256, 256, 0, stream>>>(qkv_w, proj_w, wqkv, wproj);

    k_qkv <<<12288 / 8, 256, 0, stream>>>(xb, wqkv, qkv_b, qbuf, kT, vbuf);
    k_attn<<<B_NUM * H_NUM * 4, 256, 0, stream>>>(qbuf, kT, vbuf, rel, ost);
    k_proj<<<4096 / 8, 256, 0, stream>>>(ost, wproj, proj_b, out);
}
